// NodeClustering_40948218200138
// MI455X (gfx1250) — compile-verified, matrix operand order checked by reading, not run on hardware
//
#include <hip/hip_runtime.h>

// ---------------------------------------------------------------------------
// NodeClustering for MI455X (gfx1250, wave32).
// B=8, P=16384, C=256, 8 clusters, 3 iters (reference setup constants).
//
// Design:
//  * Both 131072x256x256 GEMMs use v_wmma_f32_16x16x32_f16 (f16 in, f32 acc):
//    makes the whole pipeline memory-bound (~1.2 GB HBM traffic @ 23.3 TB/s).
//  * GEMM1 reads fp32 F_p directly (in-register f16 conversion) and fuses the
//    per-row inverse-L2-norm into its epilogue -> two full passes eliminated.
//  * n-tile loop unrolled x2 (two independent WMMA chains) + software-pipelined
//    B-fragment loads (bcur/bnext) so loads issue ahead of the consuming WMMA.
//  * Clustering argmax is invariant to the positive per-point norm scale ->
//    raw dot products; node normalization only needed in the final softmax.
//  * Final GEMM stores non-temporal (output never re-read; keep L2 for nodes).
// ---------------------------------------------------------------------------

#define B_DIM 8
#define P_DIM 16384
#define C_DIM 256
#define M_DIM (B_DIM * P_DIM)   // 131072
#define NCL 8
#define NUM_ITERS 3
#define SIM_TEMP 10.0f

typedef _Float16 h8 __attribute__((ext_vector_type(8)));
typedef _Float16 v16h __attribute__((ext_vector_type(16)));
typedef float v8f __attribute__((ext_vector_type(8)));

// ---------------------------------------------------------------------------
// Weight conversion: fp32 -> f16 (both 256x256 matrices at once)
// ---------------------------------------------------------------------------
__global__ __launch_bounds__(256)
void nc_convert_w(const float* __restrict__ pw, const float* __restrict__ rw,
                  _Float16* __restrict__ pw16, _Float16* __restrict__ rw16) {
    int i = blockIdx.x * 256 + threadIdx.x;          // 65536 elements
    pw16[i] = (_Float16)pw[i];
    rw16[i] = (_Float16)rw[i];
}

// ---------------------------------------------------------------------------
// Helper: load one f16 B fragment (16x16x32 layout) from row-major W[n][k]
// ---------------------------------------------------------------------------
__device__ __forceinline__ v16h load_bfrag(const _Float16* brow, int s) {
    h8 lo = *(const h8*)(brow + 32 * s);
    h8 hh = *(const h8*)(brow + 32 * s + 16);
    return __builtin_shufflevector(lo, hh, 0,1,2,3,4,5,6,7,8,9,10,11,12,13,14,15);
}

// Software-pipelined 8-step WMMA dot over K=256 for one 16x16 tile
__device__ __forceinline__ v8f wmma_k256(const v16h* afrag, const _Float16* brow, v8f acc) {
    v16h bcur = load_bfrag(brow, 0);
#pragma unroll
    for (int s = 0; s < 8; ++s) {
        v16h bnext;
        if (s < 7) bnext = load_bfrag(brow, s + 1);   // issue ahead of consuming WMMA
        acc = __builtin_amdgcn_wmma_f32_16x16x32_f16(
                  false, afrag[s], false, bcur, (short)0, acc, false, false);
        bcur = bnext;
    }
    return acc;
}

// ---------------------------------------------------------------------------
// GEMM1 (fused): nodes[M x 256] = F_p(fp32) @ W^T + bias, W row-major [n][k].
// Also emits inv_norm[m] = 1/max(||nodes[m]||,1e-12) from the accumulators.
// 4 waves / block; each wave owns a 16-row strip; A K-fragments preloaded
// with in-register fp32->f16 conversion.
// ---------------------------------------------------------------------------
__global__ __launch_bounds__(128)
void nc_wmma_gemm_f32a(const float* __restrict__ A, const _Float16* __restrict__ W,
                       const float* __restrict__ bias, float* __restrict__ C,
                       float* __restrict__ inv_norm) {
    const int lane = threadIdx.x & 31;
    const int wave = threadIdx.x >> 5;
    const int mbase = blockIdx.x * 64 + wave * 16;
    const int hi16 = lane >> 4;           // 0: K {0..7,16..23}; 1: K {8..15,24..31}
    const int l15  = lane & 15;
    const int row  = mbase + l15;

    // Preload 8 A fragments (K = 0..255), converting fp32 -> f16 in-register
    const float* arow = A + (size_t)row * C_DIM + hi16 * 8;
    v16h afrag[8];
#pragma unroll
    for (int s = 0; s < 8; ++s) {
        float4 q0 = *(const float4*)(arow + 32 * s);
        float4 q1 = *(const float4*)(arow + 32 * s + 4);
        float4 q2 = *(const float4*)(arow + 32 * s + 16);
        float4 q3 = *(const float4*)(arow + 32 * s + 20);
        v16h a;
        a[0]  = (_Float16)q0.x; a[1]  = (_Float16)q0.y;
        a[2]  = (_Float16)q0.z; a[3]  = (_Float16)q0.w;
        a[4]  = (_Float16)q1.x; a[5]  = (_Float16)q1.y;
        a[6]  = (_Float16)q1.z; a[7]  = (_Float16)q1.w;
        a[8]  = (_Float16)q2.x; a[9]  = (_Float16)q2.y;
        a[10] = (_Float16)q2.z; a[11] = (_Float16)q2.w;
        a[12] = (_Float16)q3.x; a[13] = (_Float16)q3.y;
        a[14] = (_Float16)q3.z; a[15] = (_Float16)q3.w;
        afrag[s] = a;
    }

    float sq[8];
#pragma unroll
    for (int v = 0; v < 8; ++v) sq[v] = 0.0f;

#pragma unroll 2
    for (int nt = 0; nt < 16; ++nt) {
        const int n = nt * 16 + l15;
        const float bv = bias[n];
        v8f acc;
#pragma unroll
        for (int i = 0; i < 8; ++i) acc[i] = bv;

        acc = wmma_k256(afrag, W + (size_t)n * C_DIM + hi16 * 8, acc);

        // D layout: element (m = v + 8*hi16, n = l15) in VGPR v
        float* crow = C + (size_t)(mbase + hi16 * 8) * C_DIM + n;
#pragma unroll
        for (int v = 0; v < 8; ++v) {
            crow[(size_t)v * C_DIM] = acc[v];
            sq[v] = fmaf(acc[v], acc[v], sq[v]);
        }
    }

    // Row-norm reduction: sum sq[v] over the 16 lanes holding that row's columns
#pragma unroll
    for (int v = 0; v < 8; ++v) {
#pragma unroll
        for (int off = 8; off; off >>= 1) sq[v] += __shfl_xor(sq[v], off, 32);
    }
    if (l15 == 0) {   // lanes 0 (rows 0..7) and 16 (rows 8..15)
#pragma unroll
        for (int v = 0; v < 8; ++v)
            inv_norm[mbase + hi16 * 8 + v] = 1.0f / fmaxf(sqrtf(sq[v]), 1e-12f);
    }
}

// ---------------------------------------------------------------------------
// GEMM2: out[M x 256] = A16(f16) @ W^T + bias. Non-temporal output stores.
// ---------------------------------------------------------------------------
__global__ __launch_bounds__(128)
void nc_wmma_gemm_f16a(const _Float16* __restrict__ A, const _Float16* __restrict__ W,
                       const float* __restrict__ bias, float* __restrict__ C) {
    const int lane = threadIdx.x & 31;
    const int wave = threadIdx.x >> 5;
    const int mbase = blockIdx.x * 64 + wave * 16;
    const int hi16 = lane >> 4;
    const int l15  = lane & 15;
    const int row  = mbase + l15;

    const _Float16* arow = A + (size_t)row * C_DIM + hi16 * 8;
    v16h afrag[8];
#pragma unroll
    for (int s = 0; s < 8; ++s) {
        h8 lo = *(const h8*)(arow + 32 * s);
        h8 hh = *(const h8*)(arow + 32 * s + 16);
        afrag[s] = __builtin_shufflevector(lo, hh, 0,1,2,3,4,5,6,7,8,9,10,11,12,13,14,15);
    }

#pragma unroll 2
    for (int nt = 0; nt < 16; ++nt) {
        const int n = nt * 16 + l15;
        const float bv = bias[n];
        v8f acc;
#pragma unroll
        for (int i = 0; i < 8; ++i) acc[i] = bv;

        acc = wmma_k256(afrag, W + (size_t)n * C_DIM + hi16 * 8, acc);

        float* crow = C + (size_t)(mbase + hi16 * 8) * C_DIM + n;
#pragma unroll
        for (int v = 0; v < 8; ++v)
            __builtin_nontemporal_store(acc[v], crow + (size_t)v * C_DIM);
    }
}

// ---------------------------------------------------------------------------
// Gather initial centers: idx_k = floor(k*(P-1)/(NCL-1)) (matches linspace+trunc)
// ---------------------------------------------------------------------------
__global__ __launch_bounds__(256)
void nc_init_centers(const float* __restrict__ nodes, float* __restrict__ centers) {
    int b = blockIdx.x >> 3, k = blockIdx.x & 7;
    long idx = (long)k * (P_DIM - 1) / (NCL - 1);
    centers[((size_t)b * NCL + k) * C_DIM + threadIdx.x] =
        nodes[((size_t)b * P_DIM + idx) * C_DIM + threadIdx.x];
}

// ---------------------------------------------------------------------------
// Normalize centers (block per center vector)
// ---------------------------------------------------------------------------
__global__ __launch_bounds__(256)
void nc_normalize_centers(const float* __restrict__ centers, float* __restrict__ cn) {
    __shared__ float red[256];
    int v = blockIdx.x;                              // b*8 + k, 64 total
    float x = centers[(size_t)v * C_DIM + threadIdx.x];
    red[threadIdx.x] = x * x;
    __syncthreads();
    for (int off = 128; off; off >>= 1) {
        if (threadIdx.x < off) red[threadIdx.x] += red[threadIdx.x + off];
        __syncthreads();
    }
    float inv = 1.0f / fmaxf(sqrtf(red[0]), 1e-12f);
    cn[(size_t)v * C_DIM + threadIdx.x] = x * inv;
}

__global__ __launch_bounds__(256)
void nc_zero(float* __restrict__ sums, float* __restrict__ counts) {
    int i = blockIdx.x * 256 + threadIdx.x;          // 16384 threads
    sums[i] = 0.0f;
    if (i < B_DIM * NCL) counts[i] = 0.0f;
}

// ---------------------------------------------------------------------------
// Assignment + center accumulation. Wave per point; centers_norm in LDS;
// LDS float atomics for partial sums, then block->global atomics.
// argmax(sim) is invariant to the positive 1/||node|| scale -> raw dots.
// ---------------------------------------------------------------------------
__global__ __launch_bounds__(256)
void nc_assign_accum(const float* __restrict__ nodes, const float* __restrict__ centers_norm,
                     float* __restrict__ sums, float* __restrict__ counts) {
    const int b = blockIdx.y;
    __shared__ float cn[NCL * C_DIM];
    __shared__ float ls[NCL * C_DIM];
    __shared__ float lc[NCL];
    for (int i = threadIdx.x; i < NCL * C_DIM; i += 256) {
        cn[i] = centers_norm[(size_t)b * NCL * C_DIM + i];
        ls[i] = 0.0f;
    }
    if (threadIdx.x < NCL) lc[threadIdx.x] = 0.0f;
    __syncthreads();

    const int lane  = threadIdx.x & 31;
    const int wid   = blockIdx.x * 8 + (threadIdx.x >> 5);
    const int nw    = gridDim.x * 8;
    const int cbase = lane * 8;
    for (int p = wid; p < P_DIM; p += nw) {
        const float* nv = nodes + ((size_t)b * P_DIM + p) * C_DIM + cbase;
        float x[8];
#pragma unroll
        for (int j = 0; j < 8; ++j) x[j] = nv[j];
        float d[NCL];
#pragma unroll
        for (int k = 0; k < NCL; ++k) {
            float t = 0.0f;
#pragma unroll
            for (int j = 0; j < 8; ++j) t = fmaf(x[j], cn[k * C_DIM + cbase + j], t);
            d[k] = t;
        }
#pragma unroll
        for (int k = 0; k < NCL; ++k)
#pragma unroll
            for (int off = 16; off; off >>= 1) d[k] += __shfl_xor(d[k], off, 32);
        int a = 0; float best = d[0];
#pragma unroll
        for (int k = 1; k < NCL; ++k) if (d[k] > best) { best = d[k]; a = k; }
#pragma unroll
        for (int j = 0; j < 8; ++j) atomicAdd(&ls[a * C_DIM + cbase + j], x[j]);
        if (lane == 0) atomicAdd(&lc[a], 1.0f);
    }
    __syncthreads();
    for (int i = threadIdx.x; i < NCL * C_DIM; i += 256)
        atomicAdd(&sums[(size_t)b * NCL * C_DIM + i], ls[i]);
    if (threadIdx.x < NCL) atomicAdd(&counts[b * NCL + threadIdx.x], lc[threadIdx.x]);
}

__global__ __launch_bounds__(256)
void nc_finalize_centers(const float* __restrict__ sums, const float* __restrict__ counts,
                         float* __restrict__ centers) {
    int v = blockIdx.x;                              // 64
    float cnt = fmaxf(counts[v], 1.0f);
    centers[(size_t)v * C_DIM + threadIdx.x] = sums[(size_t)v * C_DIM + threadIdx.x] / cnt;
}

// ---------------------------------------------------------------------------
// Final pass: sim -> softmax -> clustered + F_p, written as f16 GEMM input.
// ---------------------------------------------------------------------------
__global__ __launch_bounds__(256)
void nc_final_cluster(const float* __restrict__ nodes, const float* __restrict__ F_p,
                      const float* __restrict__ centers, const float* __restrict__ centers_norm,
                      const float* __restrict__ inv_norm, _Float16* __restrict__ A2) {
    const int b = blockIdx.y;
    __shared__ float cn[NCL * C_DIM];
    __shared__ float ce[NCL * C_DIM];
    for (int i = threadIdx.x; i < NCL * C_DIM; i += 256) {
        cn[i] = centers_norm[(size_t)b * NCL * C_DIM + i];
        ce[i] = centers[(size_t)b * NCL * C_DIM + i];
    }
    __syncthreads();

    const int lane  = threadIdx.x & 31;
    const int wid   = blockIdx.x * 8 + (threadIdx.x >> 5);
    const int nw    = gridDim.x * 8;
    const int cbase = lane * 8;
    for (int p = wid; p < P_DIM; p += nw) {
        size_t rowoff = ((size_t)b * P_DIM + p) * C_DIM + cbase;
        const float* nv = nodes + rowoff;
        float x[8];
#pragma unroll
        for (int j = 0; j < 8; ++j) x[j] = nv[j];
        float d[NCL];
#pragma unroll
        for (int k = 0; k < NCL; ++k) {
            float t = 0.0f;
#pragma unroll
            for (int j = 0; j < 8; ++j) t = fmaf(x[j], cn[k * C_DIM + cbase + j], t);
            d[k] = t;
        }
#pragma unroll
        for (int k = 0; k < NCL; ++k)
#pragma unroll
            for (int off = 16; off; off >>= 1) d[k] += __shfl_xor(d[k], off, 32);

        float inv = inv_norm[b * P_DIM + p] * SIM_TEMP;
        float m = d[0] * inv;
#pragma unroll
        for (int k = 1; k < NCL; ++k) m = fmaxf(m, d[k] * inv);
        float w[NCL], wsum = 0.0f;
#pragma unroll
        for (int k = 0; k < NCL; ++k) { w[k] = __expf(d[k] * inv - m); wsum += w[k]; }
        float r = 1.0f / wsum;
#pragma unroll
        for (int k = 0; k < NCL; ++k) w[k] *= r;

#pragma unroll
        for (int j = 0; j < 8; ++j) {
            float acc = F_p[rowoff + j];
#pragma unroll
            for (int k = 0; k < NCL; ++k) acc = fmaf(w[k], ce[k * C_DIM + cbase + j], acc);
            A2[rowoff + j] = (_Float16)acc;
        }
    }
}

// ---------------------------------------------------------------------------
// Host-side launcher
// ---------------------------------------------------------------------------
extern "C" void kernel_launch(void* const* d_in, const int* in_sizes, int n_in,
                              void* d_out, int out_size, void* d_ws, size_t ws_size,
                              hipStream_t stream) {
    (void)in_sizes; (void)n_in; (void)out_size; (void)ws_size;
    const float* F_p      = (const float*)d_in[0];
    const float* proj_w   = (const float*)d_in[1];
    const float* proj_b   = (const float*)d_in[2];
    const float* refine_w = (const float*)d_in[3];
    const float* refine_b = (const float*)d_in[4];
    float* out = (float*)d_out;

    // Workspace carve-up (256B-aligned slabs)
    char* ws = (char*)d_ws;
    size_t off = 0;
    auto carve = [&](size_t bytes) -> char* {
        char* p = ws + off;
        off += (bytes + 255) & ~(size_t)255;
        return p;
    };
    float*    nodes    = (float*)   carve((size_t)M_DIM * C_DIM * 4);   // 134 MB
    _Float16* A16      = (_Float16*)carve((size_t)M_DIM * C_DIM * 2);   //  67 MB
    _Float16* pw16     = (_Float16*)carve((size_t)C_DIM * C_DIM * 2);
    _Float16* rw16     = (_Float16*)carve((size_t)C_DIM * C_DIM * 2);
    float*    inv_norm = (float*)   carve((size_t)M_DIM * 4);
    float*    centers  = (float*)   carve((size_t)B_DIM * NCL * C_DIM * 4);
    float*    cnorm    = (float*)   carve((size_t)B_DIM * NCL * C_DIM * 4);
    float*    sums     = (float*)   carve((size_t)B_DIM * NCL * C_DIM * 4);
    float*    counts   = (float*)   carve((size_t)B_DIM * NCL * 4);

    // 1) weight f16 conversion (tiny)
    nc_convert_w<<<(C_DIM * C_DIM) / 256, 256, 0, stream>>>(proj_w, refine_w, pw16, rw16);

    // 2) nodes = F_p @ proj_w^T + proj_b (WMMA, fused fp32->f16 A conversion
    //    and fused per-row inverse norms)
    nc_wmma_gemm_f32a<<<M_DIM / 64, 128, 0, stream>>>(F_p, pw16, proj_b, nodes, inv_norm);

    // 3) clustering (argmax is scale-invariant -> raw node/center dots)
    nc_init_centers<<<B_DIM * NCL, 256, 0, stream>>>(nodes, centers);
    for (int it = 0; it < NUM_ITERS; ++it) {
        nc_normalize_centers<<<B_DIM * NCL, 256, 0, stream>>>(centers, cnorm);
        nc_zero<<<(B_DIM * NCL * C_DIM) / 256, 256, 0, stream>>>(sums, counts);
        nc_assign_accum<<<dim3(64, B_DIM), 256, 0, stream>>>(nodes, cnorm, sums, counts);
        nc_finalize_centers<<<B_DIM * NCL, 256, 0, stream>>>(sums, counts, centers);
    }
    nc_normalize_centers<<<B_DIM * NCL, 256, 0, stream>>>(centers, cnorm);

    // 4) softmax blend + residual, emitted directly as f16 GEMM input
    nc_final_cluster<<<dim3(64, B_DIM), 256, 0, stream>>>(nodes, F_p, centers, cnorm,
                                                          inv_norm, A16);

    // 5) out = (clustered + F_p) @ refine_w^T + refine_b (WMMA, NT stores)
    nc_wmma_gemm_f16a<<<M_DIM / 64, 128, 0, stream>>>(A16, rw16, refine_b, out);
}
